// RotaryAttention_3693671874658
// MI455X (gfx1250) — matrix operand order repaired and checked by reading, not compiled, not run on hardware
//
#include <hip/hip_runtime.h>
#include <hip/hip_bf16.h>

// ---------------------------------------------------------------------------
// RotaryAttention on MI455X (gfx1250, wave32, WMMA + async global->LDS)
//   B=2, T=S=256, E=512, H=8, D=64
// ---------------------------------------------------------------------------

#define BB  2
#define TT  256
#define SS  256
#define EE  512
#define HH  8
#define DD  64
#define BT  (BB * TT)          // 512 rows of x
#define N_QKV (3 * EE)         // 1536

typedef __attribute__((ext_vector_type(16))) __bf16 bf16x16;
typedef __attribute__((ext_vector_type(8)))  __bf16 bf16x8;
typedef __attribute__((ext_vector_type(8)))  float  f32x8;

// ---------------------------------------------------------------------------
// CDNA5 async global->LDS copy (ASYNCcnt path), per-lane 16B.
//   vdst VGPR = LDS byte address, vaddr = 64-bit global address.
// ---------------------------------------------------------------------------
__device__ __forceinline__ void async_copy_b128(unsigned lds_off,
                                                unsigned long long gaddr) {
  asm volatile("global_load_async_to_lds_b128 %0, %1, off"
               :: "v"(lds_off), "v"(gaddr) : "memory");
}
#define WAIT_ASYNCCNT(n) asm volatile("s_wait_asynccnt " #n ::: "memory")

// ---------------------------------------------------------------------------
// WMMA fragment helpers (wave32; A/B k order is this code's contiguous choice, not the ISA table):
//   A frag elem e (lane): row = lane&15, K = (lane>>4)*16 + e
//   B frag elem e (lane): col = lane&15, K = (lane>>4)*16 + e
//   C/D  elem i (lane): row = i + 8*(lane>>4), col = lane&15
// ---------------------------------------------------------------------------
__device__ __forceinline__ bf16x16 load_a_frag(const __bf16* __restrict__ p) {
  bf16x8 lo = *(const bf16x8*)p;
  bf16x8 hi = *(const bf16x8*)(p + 8);
  return __builtin_shufflevector(lo, hi,
           0, 1, 2, 3, 4, 5, 6, 7, 8, 9, 10, 11, 12, 13, 14, 15);
}

// 32x32 macro-tile per wave (2x2 WMMA tiles): 8 b128 loads feed 4 WMMAs.
__device__ __forceinline__ void gemm_bf16_2x2(const __bf16* __restrict__ A, int lda,
                                              const __bf16* __restrict__ Bt, int ldb,
                                              int rowBase, int colBase, int K,
                                              int lane, f32x8 c[2][2]) {
  const f32x8 z = {0.f, 0.f, 0.f, 0.f, 0.f, 0.f, 0.f, 0.f};
  c[0][0] = z; c[0][1] = z; c[1][0] = z; c[1][1] = z;
  const int m    = lane & 15;
  const int half = lane >> 4;
  const __bf16* ap = A  + (size_t)(rowBase + m) * lda + half * 16;
  const __bf16* bp = Bt + (size_t)(colBase + m) * ldb + half * 16;
  const size_t astep = (size_t)16 * lda;
  const size_t bstep = (size_t)16 * ldb;
#pragma unroll 4
  for (int k0 = 0; k0 < K; k0 += 32) {
    bf16x16 a0 = load_a_frag(ap + k0);
    bf16x16 a1 = load_a_frag(ap + astep + k0);
    bf16x16 b0 = *(const bf16x16*)(bp + k0);
    bf16x16 b1 = *(const bf16x16*)(bp + bstep + k0);
    c[0][0] = __builtin_amdgcn_wmma_f32_16x16x32_bf16(false, a0, false, b0,
                (short)0, c[0][0], false, false);
    c[0][1] = __builtin_amdgcn_wmma_f32_16x16x32_bf16(false, a0, false, b1,
                (short)0, c[0][1], false, false);
    c[1][0] = __builtin_amdgcn_wmma_f32_16x16x32_bf16(false, a1, false, b0,
                (short)0, c[1][0], false, false);
    c[1][1] = __builtin_amdgcn_wmma_f32_16x16x32_bf16(false, a1, false, b1,
                (short)0, c[1][1], false, false);
  }
}

// ---------------------------------------------------------------------------
// Converters (fp32 -> bf16; weights transposed so WMMA B-frags are contiguous
// 32-byte per-lane loads)
// ---------------------------------------------------------------------------
__global__ void cvt_bf16_kernel(const float* __restrict__ in,
                                __bf16* __restrict__ out, int n) {
  int i = blockIdx.x * blockDim.x + threadIdx.x;
  if (i < n) out[i] = (__bf16)in[i];
}

// in: [rows(K) x cols(N)] fp32  ->  out: [cols(N) x rows(K)] bf16
__global__ void cvt_bf16_transpose_kernel(const float* __restrict__ in,
                                          __bf16* __restrict__ out,
                                          int rows, int cols) {
  int i = blockIdx.x * blockDim.x + threadIdx.x;
  if (i < rows * cols) {
    int n = i / rows;
    int k = i - n * rows;
    out[i] = (__bf16)in[(size_t)k * cols + n];
  }
}

// ---------------------------------------------------------------------------
// K1: qkv = x @ Wqkv + bqkv, scattered into per-head layouts:
//   q -> fp32 (b,h,t,d); k -> fp32 (b,h,d,s); v -> bf16 (b,h,d,s)
// ---------------------------------------------------------------------------
__global__ __launch_bounds__(128)
void qkv_gemm_kernel(const __bf16* __restrict__ xbf,
                     const __bf16* __restrict__ wqkvt,
                     const float*  __restrict__ bqkv,
                     float* __restrict__ qf,
                     float* __restrict__ ktf,
                     __bf16* __restrict__ vtbf) {
  const int lane = threadIdx.x & 31;
  const int wave = threadIdx.x >> 5;
  const int nt2  = blockIdx.x * 4 + wave;   // 0..47  (32-wide col blocks)
  const int mt2  = blockIdx.y;              // 0..15  (32-wide row blocks)

  f32x8 c[2][2];
  gemm_bf16_2x2(xbf, EE, wqkvt, EE, mt2 * 32, nt2 * 32, EE, lane, c);

  const int half = lane >> 4;
#pragma unroll
  for (int nn = 0; nn < 2; nn++) {
    const int n    = nt2 * 32 + nn * 16 + (lane & 15);   // [0,1536)
    const float bias = bqkv[n];
    const int sec = n >> 9;                               // 0:q 1:k 2:v
    const int rem = n & 511;
    const int h   = rem >> 6;
    const int d   = rem & 63;
#pragma unroll
    for (int mm = 0; mm < 2; mm++) {
#pragma unroll
      for (int i = 0; i < 8; i++) {
        int r = mt2 * 32 + mm * 16 + i + 8 * half;        // [0,512)
        int b = r >> 8, t = r & 255;
        int bh = b * HH + h;
        float val = c[mm][nn][i] + bias;
        if (sec == 0)      qf  [((bh * TT + t) << 6) + d] = val;
        else if (sec == 1) ktf [((bh * DD + d) << 8) + t] = val;
        else               vtbf[((bh * DD + d) << 8) + t] = (__bf16)val;
      }
    }
  }
}

// ---------------------------------------------------------------------------
// K2: fused rotary scores + softmax. One block per (b,t); thread = s.
// The 33.5 MB mask stream (the HBM roofline term) is staged via CDNA5 async
// global->LDS b128 copies in a 4-chunk / 2-buffer pipeline so copies of
// chunk N+1 overlap FMA work on chunk N. Heads amortized via:
//   q . (k (*) m) = m0*(q0*k0 + q1*k1) + m1*(q1*k0 - q0*k1)
// ---------------------------------------------------------------------------
#define CW 20   // LDS words per row per chunk (16 data + 4 pad, 16B aligned)

__global__ __launch_bounds__(256)
void score_softmax_kernel(const float* __restrict__ qf,
                          const float* __restrict__ ktf,
                          const float* __restrict__ mask,
                          const float* __restrict__ pad,
                          __bf16* __restrict__ pbf,
                          float* __restrict__ attn_w_out) {
  const int bt = blockIdx.x;
  const int b  = bt >> 8;
  const int t  = bt & 255;
  const int s  = threadIdx.x;

  __shared__ float qs[HH * DD];          // 2 KB
  __shared__ float red[SS];              // 1 KB
  __shared__ float sA[HH * SS];          // 8 KB
  __shared__ float sM[2][SS * CW];       // 2 x 20 KB double-buffered mask

  const float* gmask = mask + ((size_t)bt << 14);   // bt * 256 * 64 floats
  const unsigned smb[2] = {
    (unsigned)(size_t)(void*)&sM[0][0],
    (unsigned)(size_t)(void*)&sM[1][0]
  };

  // stage(): copy mask chunk cc (16 words of every row) into buffer `buf`.
  // 4 async b128 per thread; 64B-contiguous segments per 4 lanes.
  const int cpiece = threadIdx.x & 3;          // 16B piece within chunk
  const int rbase  = threadIdx.x >> 2;         // 0..63
#define STAGE(cc, buf)                                                        \
  do {                                                                        \
    _Pragma("unroll")                                                         \
    for (int j = 0; j < 4; j++) {                                             \
      int srow = rbase + 64 * j;                                              \
      const float* g = gmask + srow * 64 + (cc) * 16 + cpiece * 4;            \
      unsigned l = smb[buf] + (unsigned)((srow * CW + cpiece * 4) * 4);       \
      async_copy_b128(l, (unsigned long long)(const void*)g);                 \
    }                                                                         \
  } while (0)

  STAGE(0, 0);                          // 4 in flight
  STAGE(1, 1);                          // 8 in flight

  // overlap: load q for all heads while the copies fly
  for (int j = threadIdx.x; j < HH * DD; j += 256) {
    int h = j >> 6, d = j & 63;
    qs[j] = qf[(((b * HH + h) * TT + t) << 6) + d];
  }

  float acc[HH];
#pragma unroll
  for (int h = 0; h < HH; h++) acc[h] = 0.f;
  const float* kb = ktf + (size_t)b * HH * DD * SS;

  // compute 8 p-pairs of chunk cc out of buffer `buf`
#define COMPUTE(cc, buf)                                                      \
  do {                                                                        \
    const float* mr = &sM[buf][s * CW];                                       \
    _Pragma("unroll")                                                         \
    for (int pp = 0; pp < 8; pp++) {                                          \
      int p = (cc) * 8 + pp;                                                  \
      float m0 = mr[2 * pp];                                                  \
      float m1 = mr[2 * pp + 1];                                              \
      _Pragma("unroll")                                                       \
      for (int h = 0; h < HH; h++) {                                          \
        float q0 = qs[h * DD + 2 * p];                                        \
        float q1 = qs[h * DD + 2 * p + 1];                                    \
        float k0 = kb[((h * DD + 2 * p) << 8) + s];                           \
        float k1 = kb[((h * DD + 2 * p + 1) << 8) + s];                       \
        acc[h] += m0 * (q0 * k0 + q1 * k1) + m1 * (q1 * k0 - q0 * k1);        \
      }                                                                       \
    }                                                                         \
  } while (0)

  WAIT_ASYNCCNT(0x4);  __syncthreads();          // chunk0 ready everywhere
  COMPUTE(0, 0);
  __syncthreads();  STAGE(2, 0);                 // buf0 free -> prefetch chunk2
  WAIT_ASYNCCNT(0x4);  __syncthreads();          // chunk1 ready
  COMPUTE(1, 1);
  __syncthreads();  STAGE(3, 1);                 // buf1 free -> prefetch chunk3
  WAIT_ASYNCCNT(0x4);  __syncthreads();          // chunk2 ready
  COMPUTE(2, 0);
  WAIT_ASYNCCNT(0x0);  __syncthreads();          // chunk3 ready
  COMPUTE(3, 1);

  const float scale = 0.125f;                    // 1/sqrt(64)
  const float padv  = pad[b * TT + s];           // pad_mask (B,1,1,T)
#pragma unroll
  for (int h = 0; h < HH; h++) sA[h * SS + s] = acc[h] * scale + padv;
  __syncthreads();

  float wsum = 0.f;
  for (int h = 0; h < HH; h++) {
    float v = sA[h * SS + s];
    red[s] = v;
    __syncthreads();
    for (int off = 128; off > 0; off >>= 1) {
      if (s < off) red[s] = fmaxf(red[s], red[s + off]);
      __syncthreads();
    }
    float mx = red[0];
    __syncthreads();
    float e = __expf(v - mx);
    red[s] = e;
    __syncthreads();
    for (int off = 128; off > 0; off >>= 1) {
      if (s < off) red[s] += red[s + off];
      __syncthreads();
    }
    float pr = e * (1.0f / red[0]);
    __syncthreads();
    pbf[(((size_t)(b * HH + h) * TT + t) << 8) + s] = (__bf16)pr;
    wsum += pr;
  }
  attn_w_out[((size_t)bt << 8) + s] = wsum * (1.0f / HH);
}

// ---------------------------------------------------------------------------
// K3: O = P @ V per (b,h). V stored transposed bf16 (d,s) == WMMA B source.
// Context written bf16 (b,t,E) as the out-projection's A matrix.
// ---------------------------------------------------------------------------
__global__ __launch_bounds__(64)
void pv_gemm_kernel(const __bf16* __restrict__ pbf,
                    const __bf16* __restrict__ vtbf,
                    __bf16* __restrict__ octx) {
  const int lane = threadIdx.x & 31;
  const int nt2  = threadIdx.x >> 5;     // 0..1 (32-wide d blocks)
  const int mt2  = blockIdx.x;           // 0..7 (32-wide t blocks)
  const int bh   = blockIdx.y;           // 0..15

  const __bf16* A  = pbf  + (size_t)bh * TT * SS;
  const __bf16* Bt = vtbf + (size_t)bh * DD * SS;
  f32x8 c[2][2];
  gemm_bf16_2x2(A, SS, Bt, SS, mt2 * 32, nt2 * 32, SS, lane, c);

  const int b    = bh >> 3;
  const int h    = bh & 7;
  const int half = lane >> 4;
#pragma unroll
  for (int nn = 0; nn < 2; nn++) {
    int d = nt2 * 32 + nn * 16 + (lane & 15);
#pragma unroll
    for (int mm = 0; mm < 2; mm++) {
#pragma unroll
      for (int i = 0; i < 8; i++) {
        int t = mt2 * 32 + mm * 16 + i + 8 * half;
        octx[(((size_t)(b * TT + t)) << 9) + h * DD + d] = (__bf16)c[mm][nn][i];
      }
    }
  }
}

// ---------------------------------------------------------------------------
// K4: out = octx @ Wo + bo  -> d_out (B,T,E) fp32
// ---------------------------------------------------------------------------
__global__ __launch_bounds__(128)
void oproj_gemm_kernel(const __bf16* __restrict__ octx,
                       const __bf16* __restrict__ wot,
                       const float* __restrict__ bo,
                       float* __restrict__ out) {
  const int lane = threadIdx.x & 31;
  const int wave = threadIdx.x >> 5;
  const int nt2  = blockIdx.x * 4 + wave;   // 0..15
  const int mt2  = blockIdx.y;              // 0..15

  f32x8 c[2][2];
  gemm_bf16_2x2(octx, EE, wot, EE, mt2 * 32, nt2 * 32, EE, lane, c);

  const int half = lane >> 4;
#pragma unroll
  for (int nn = 0; nn < 2; nn++) {
    int n = nt2 * 32 + nn * 16 + (lane & 15);
    float bias = bo[n];
#pragma unroll
    for (int mm = 0; mm < 2; mm++) {
#pragma unroll
      for (int i = 0; i < 8; i++) {
        int r = mt2 * 32 + mm * 16 + i + 8 * half;
        out[((size_t)r << 9) + n] = c[mm][nn][i] + bias;
      }
    }
  }
}

// ---------------------------------------------------------------------------
// Host-side launch
// inputs: 0:x 1:attn_mask 2:pad_mask 3:Wqkv 4:bqkv 5:Wo 6:bo
// d_out: o (B*T*E fp32) ++ attn_weights (B*T*S fp32)
// ---------------------------------------------------------------------------
extern "C" void kernel_launch(void* const* d_in, const int* in_sizes, int n_in,
                              void* d_out, int out_size, void* d_ws, size_t ws_size,
                              hipStream_t stream) {
  const float* x     = (const float*)d_in[0];
  const float* amask = (const float*)d_in[1];
  const float* pmask = (const float*)d_in[2];
  const float* Wqkv  = (const float*)d_in[3];
  const float* bqkv  = (const float*)d_in[4];
  const float* Wo    = (const float*)d_in[5];
  const float* bo    = (const float*)d_in[6];
  float* out_o = (float*)d_out;
  float* out_w = (float*)d_out + (size_t)BB * TT * EE;

  char* ws = (char*)d_ws;
  size_t off = 0;
  __bf16* xbf   = (__bf16*)(ws + off); off += (size_t)BT * EE * 2;           // 512 KB
  __bf16* wqkvt = (__bf16*)(ws + off); off += (size_t)N_QKV * EE * 2;        // 1.5 MB
  __bf16* wot   = (__bf16*)(ws + off); off += (size_t)EE * EE * 2;           // 512 KB
  float*  qf    = (float*)(ws + off);  off += (size_t)BB * HH * TT * DD * 4; // 1 MB
  float*  ktf   = (float*)(ws + off);  off += (size_t)BB * HH * DD * SS * 4; // 1 MB
  __bf16* vtbf  = (__bf16*)(ws + off); off += (size_t)BB * HH * DD * SS * 2; // 512 KB
  __bf16* pbf   = (__bf16*)(ws + off); off += (size_t)BB * HH * TT * SS * 2; // 2 MB
  __bf16* octx  = (__bf16*)(ws + off); off += (size_t)BT * EE * 2;           // 512 KB
  (void)ws_size; (void)in_sizes; (void)n_in; (void)out_size;

  {
    int n = BT * EE;
    cvt_bf16_kernel<<<(n + 255) / 256, 256, 0, stream>>>(x, xbf, n);
  }
  {
    int n = EE * N_QKV;
    cvt_bf16_transpose_kernel<<<(n + 255) / 256, 256, 0, stream>>>(Wqkv, wqkvt, EE, N_QKV);
  }
  {
    int n = EE * EE;
    cvt_bf16_transpose_kernel<<<(n + 255) / 256, 256, 0, stream>>>(Wo, wot, EE, EE);
  }

  // QKV projection (WMMA 2x2 macro-tiles) with fused head-layout scatter
  qkv_gemm_kernel<<<dim3(N_QKV / 32 / 4, BT / 32), 128, 0, stream>>>(
      xbf, wqkvt, bqkv, qf, ktf, vtbf);

  // fused rotary scores + softmax + attn_weights (async-LDS mask pipeline)
  score_softmax_kernel<<<BB * TT, SS, 0, stream>>>(
      qf, ktf, amask, pmask, pbf, out_w);

  // P @ V (WMMA)
  pv_gemm_kernel<<<dim3(TT / 32, BB * HH), 64, 0, stream>>>(pbf, vtbf, octx);

  // output projection (WMMA)
  oproj_gemm_kernel<<<dim3(EE / 32 / 4, BT / 32), 128, 0, stream>>>(
      octx, wot, bo, out_o);
}
